// VectorQuantizer_8778913153325
// MI455X (gfx1250) — matrix operand order repaired and checked by reading, not compiled, not run on hardware
//
#include <hip/hip_runtime.h>
#include <hip/hip_bf16.h>
#include <math.h>

// ---------------------------------------------------------------------------
// VQ-VAE vector quantizer, MI455X (gfx1250, wave32).
//  - GEMMs on v_wmma_f32_16x16x32_bf16 (f32 accumulate)
//  - codebook tiles staged with GLOBAL_LOAD_ASYNC_TO_LDS_B128 (ASYNCcnt),
//    double-buffered LDS, one barrier + one s_wait_asynccnt per chunk
//  - each wave owns 2 M-tiles: every B operand feeds 2 WMMAs (half the LDS
//    traffic per FLOP) and gives 2 independent accumulation chains
// ---------------------------------------------------------------------------

typedef __bf16 bf16_t;
typedef __attribute__((ext_vector_type(16))) __bf16 v16bf;
typedef __attribute__((ext_vector_type(8)))  float  v8f;

#define VQ_K     4096      // codebook size
#define VQ_D     128       // latent dim
#define VQ_N     16384     // 16 * 1024 tokens
#define VQ_ND    (VQ_N * VQ_D)        // 2097152
#define VQ_SCAL  (VQ_ND + VQ_N)       // scalar region base in d_out
#define VQ_PITCH 17                   // LDS row pitch in uint4 (bank rotation)

union V16 { uint4 u[2]; v16bf v; };

// ---------------- device-global scratch (avoids ws_size assumptions) -------
__device__ bf16_t   g_cb_bf[VQ_K * VQ_D];    // 1 MB
__device__ bf16_t   g_x_bf [VQ_N * VQ_D];    // 4 MB
__device__ float    g_sqc  [VQ_K];
__device__ float    g_invc [VQ_K];
__device__ float    g_invx [VQ_N];
__device__ int      g_idx  [VQ_N];
__device__ unsigned g_counts[VQ_K];
__device__ float    g_selcos;
__device__ float    g_mse;
__device__ float    g_dsum;
__device__ unsigned g_dmin;

// ---------------------------------------------------------------------------
// Async staging helpers (CDNA5 ASYNCcnt path).
// For a flat pointer into __shared__, addr[31:0] is the LDS byte offset
// (ISA 10.2 aperture mapping), which is exactly what VDST of
// global_load_async_to_lds_* expects.
// ---------------------------------------------------------------------------
__device__ __forceinline__ void vq_wait_async0() {
  asm volatile("s_wait_asynccnt 0x0" ::: "memory");
}

// copy one 64-code chunk (64 x 256B) into LDS with pitch VQ_PITCH uint4
__device__ __forceinline__ void vq_stage_async(uint4* dst, const bf16_t* gsrc,
                                               int tid) {
  const unsigned long long base = (unsigned long long)(uintptr_t)gsrc;
  #pragma unroll
  for (int k = 0; k < 8; ++k) {
    const int i = tid + k * 128;                       // 0..1023 uint4
    const unsigned lds = (unsigned)(uintptr_t)(dst + ((i >> 4) * VQ_PITCH + (i & 15)));
    const unsigned go  = (unsigned)(i * 16);           // byte offset from base
    asm volatile("global_load_async_to_lds_b128 %0, %1, %2"
                 :: "v"(lds), "v"(go), "s"(base) : "memory");
  }
}

// ---------------------------------------------------------------------------
__global__ __launch_bounds__(256) void vq_init() {
  int t = blockIdx.x * 256 + threadIdx.x;
  if (t < VQ_K) g_counts[t] = 0u;
  if (t == 0) { g_selcos = 0.f; g_mse = 0.f; g_dsum = 0.f; g_dmin = 0x7f800000u; }
}

// one block (128 thr) per row: bf16 copy + norm stats
__global__ __launch_bounds__(128) void vq_prep_cb(const float* __restrict__ src) {
  __shared__ float s[128];
  const int row = blockIdx.x, t = threadIdx.x;
  float v = src[row * VQ_D + t];
  g_cb_bf[row * VQ_D + t] = (bf16_t)v;
  s[t] = v * v;
  __syncthreads();
  for (int off = 64; off; off >>= 1) { if (t < off) s[t] += s[t + off]; __syncthreads(); }
  if (t == 0) {
    float q = s[0];
    g_sqc[row]  = q;
    g_invc[row] = 1.0f / fmaxf(sqrtf(q), 1e-12f);
  }
}

__global__ __launch_bounds__(128) void vq_prep_x(const float* __restrict__ src) {
  __shared__ float s[128];
  const int row = blockIdx.x, t = threadIdx.x;
  float v = src[row * VQ_D + t];
  g_x_bf[row * VQ_D + t] = (bf16_t)v;
  s[t] = v * v;
  __syncthreads();
  for (int off = 64; off; off >>= 1) { if (t < off) s[t] += s[t + off]; __syncthreads(); }
  if (t == 0) g_invx[row] = 1.0f / fmaxf(sqrtf(s[0]), 1e-12f);
}

// ---------------------------------------------------------------------------
// scores: grid 128 x 128thr (4 waves). Wave owns 32 token rows = 2 M-tiles
// (A resident in VGPRs). Loop over 4096 codes in 64-code chunks staged
// asynchronously into double-buffered LDS. Epilogue: per-row argmax of
// (xc - 0.5*||c||^2)  ==  argmin of squared euclidean distance.
// ---------------------------------------------------------------------------
__global__ __launch_bounds__(128) void vq_scores(float* __restrict__ out_idx) {
  __shared__ uint4 sB[2][64 * VQ_PITCH];
  const int tid  = threadIdx.x;
  const int lane = tid & 31;
  const int wave = tid >> 5;
  const int l15  = lane & 15;
  const int h    = lane >> 4;
  const int rowBase = blockIdx.x * 128 + wave * 32;

  // A operands for 2 tiles x 4 K-steps
  V16 tmp;
  v16bf A[2][4];
  #pragma unroll
  for (int t = 0; t < 2; ++t) {
    const uint4* xrow = (const uint4*)(g_x_bf) + (rowBase + t * 16 + l15) * 16;
    #pragma unroll
    for (int kb = 0; kb < 4; ++kb) {
      tmp.u[0] = xrow[4 * kb + h];        // K = 32kb + 8h    .. +8
      tmp.u[1] = xrow[4 * kb + h + 2];    // K = 32kb + 16+8h .. +8
      A[t][kb] = tmp.v;
    }
  }

  float best[2][8], bxc[2][8]; int bidx[2][8];
  #pragma unroll
  for (int t = 0; t < 2; ++t)
    #pragma unroll
    for (int r = 0; r < 8; ++r) { best[t][r] = -3.4e38f; bxc[t][r] = 0.f; bidx[t][r] = 0; }

  vq_stage_async(sB[0], g_cb_bf, tid);                 // prefetch chunk 0

  for (int ch = 0; ch < 64; ++ch) {
    const int cur = ch & 1;
    vq_wait_async0();                                  // my copies for chunk ch done
    __syncthreads();                                   // everyone's done; prev buf free
    if (ch + 1 < 64)
      vq_stage_async(sB[cur ^ 1], g_cb_bf + (ch + 1) * 64 * VQ_D, tid);

    const int c0 = ch * 64;
    #pragma unroll
    for (int sub = 0; sub < 4; ++sub) {
      const int ccol = c0 + sub * 16 + l15;            // this lane's column (code)
      const float sc = g_sqc[ccol];
      const uint4* brow = sB[cur] + (sub * 16 + l15) * VQ_PITCH;
      v8f acc0 = {0.f,0.f,0.f,0.f,0.f,0.f,0.f,0.f};
      v8f acc1 = {0.f,0.f,0.f,0.f,0.f,0.f,0.f,0.f};
      #pragma unroll
      for (int kb = 0; kb < 4; ++kb) {
        V16 tb;
        tb.u[0] = brow[4 * kb + h];                    // K = 32kb+8h    .. +8
        tb.u[1] = brow[4 * kb + h + 2];                // K = 32kb+16+8h .. +8
        acc0 = __builtin_amdgcn_wmma_f32_16x16x32_bf16(
            false, A[0][kb], false, tb.v, (short)0, acc0, false, false);
        acc1 = __builtin_amdgcn_wmma_f32_16x16x32_bf16(
            false, A[1][kb], false, tb.v, (short)0, acc1, false, false);
      }
      #pragma unroll
      for (int r = 0; r < 8; ++r) {                    // row = rowBase + t*16 + r + 8h
        float x0 = acc0[r], s0 = x0 - 0.5f * sc;
        if (s0 > best[0][r]) { best[0][r] = s0; bidx[0][r] = ccol; bxc[0][r] = x0; }
        float x1 = acc1[r], s1 = x1 - 0.5f * sc;
        if (s1 > best[1][r]) { best[1][r] = s1; bidx[1][r] = ccol; bxc[1][r] = x1; }
      }
    }
  }

  // per-row argmax across the 16 lanes of each half-wave (rows never mix)
  #pragma unroll
  for (int mask = 8; mask >= 1; mask >>= 1) {
    #pragma unroll
    for (int t = 0; t < 2; ++t)
      #pragma unroll
      for (int r = 0; r < 8; ++r) {
        float ob = __shfl_xor(best[t][r], mask, 32);
        int   oi = __shfl_xor(bidx[t][r], mask, 32);
        float ox = __shfl_xor(bxc[t][r],  mask, 32);
        if (ob > best[t][r] || (ob == best[t][r] && oi < bidx[t][r])) {
          best[t][r] = ob; bidx[t][r] = oi; bxc[t][r] = ox;
        }
      }
  }

  if (l15 == 0) {
    float csum = 0.f;
    #pragma unroll
    for (int t = 0; t < 2; ++t)
      #pragma unroll
      for (int r = 0; r < 8; ++r) {
        const int row = rowBase + t * 16 + r + 8 * h;
        const int bi  = bidx[t][r];
        g_idx[row]   = bi;
        out_idx[row] = (float)bi;
        atomicAdd(&g_counts[bi], 1u);
        csum += bxc[t][r] * g_invx[row] * g_invc[bi];  // cos at selected code
      }
    atomicAdd(&g_selcos, csum);
  }
}

// ---------------------------------------------------------------------------
// gram: codebook x codebook^T. grid 256 = 32 i-blocks(128 rows) x 8 j-splits.
// d = sqrt(max(sq_i + sq_j - 2*G, 0)); off-diagonal sum + min.
// ---------------------------------------------------------------------------
__global__ __launch_bounds__(128) void vq_gram() {
  __shared__ uint4 sB[2][64 * VQ_PITCH];
  __shared__ float ssum[128];
  const int tid  = threadIdx.x;
  const int lane = tid & 31;
  const int wave = tid >> 5;
  const int l15  = lane & 15;
  const int h    = lane >> 4;
  const int ib   = blockIdx.x >> 3;                    // 0..31
  const int js   = blockIdx.x & 7;                     // 0..7
  const int iBase = ib * 128 + wave * 32;

  V16 tmp;
  v16bf A[2][4];
  float sqi[2][8];
  #pragma unroll
  for (int t = 0; t < 2; ++t) {
    const uint4* arow = (const uint4*)(g_cb_bf) + (iBase + t * 16 + l15) * 16;
    #pragma unroll
    for (int kb = 0; kb < 4; ++kb) {
      tmp.u[0] = arow[4 * kb + h];
      tmp.u[1] = arow[4 * kb + h + 2];
      A[t][kb] = tmp.v;
    }
    #pragma unroll
    for (int r = 0; r < 8; ++r) sqi[t][r] = g_sqc[iBase + t * 16 + r + 8 * h];
  }

  float lsum = 0.f, lmin = 3.4e38f;

  const int jc0 = js * 8;                              // 8 chunks of 64 codes
  vq_stage_async(sB[0], g_cb_bf + jc0 * 64 * VQ_D, tid);

  for (int jj = 0; jj < 8; ++jj) {
    const int cur = jj & 1;
    vq_wait_async0();
    __syncthreads();
    if (jj + 1 < 8)
      vq_stage_async(sB[cur ^ 1], g_cb_bf + (jc0 + jj + 1) * 64 * VQ_D, tid);

    const int c0 = (jc0 + jj) * 64;
    #pragma unroll
    for (int sub = 0; sub < 4; ++sub) {
      const int ccol = c0 + sub * 16 + l15;
      const float scj = g_sqc[ccol];
      const uint4* brow = sB[cur] + (sub * 16 + l15) * VQ_PITCH;
      v8f acc0 = {0.f,0.f,0.f,0.f,0.f,0.f,0.f,0.f};
      v8f acc1 = {0.f,0.f,0.f,0.f,0.f,0.f,0.f,0.f};
      #pragma unroll
      for (int kb = 0; kb < 4; ++kb) {
        V16 tb;
        tb.u[0] = brow[4 * kb + h];
        tb.u[1] = brow[4 * kb + h + 2];
        acc0 = __builtin_amdgcn_wmma_f32_16x16x32_bf16(
            false, A[0][kb], false, tb.v, (short)0, acc0, false, false);
        acc1 = __builtin_amdgcn_wmma_f32_16x16x32_bf16(
            false, A[1][kb], false, tb.v, (short)0, acc1, false, false);
      }
      #pragma unroll
      for (int r = 0; r < 8; ++r) {
        const int r0 = iBase + r + 8 * h;
        float d2a = sqi[0][r] + scj - 2.0f * acc0[r];
        float dda = sqrtf(fmaxf(d2a, 0.f));
        if (r0 != ccol) { lsum += dda; lmin = fminf(lmin, dda); }
        const int r1 = iBase + 16 + r + 8 * h;
        float d2b = sqi[1][r] + scj - 2.0f * acc1[r];
        float ddb = sqrtf(fmaxf(d2b, 0.f));
        if (r1 != ccol) { lsum += ddb; lmin = fminf(lmin, ddb); }
      }
    }
  }

  ssum[tid] = lsum;
  __syncthreads();
  for (int off = 64; off; off >>= 1) { if (tid < off) ssum[tid] += ssum[tid + off]; __syncthreads(); }
  if (tid == 0) atomicAdd(&g_dsum, ssum[0]);
  atomicMin(&g_dmin, __float_as_uint(lmin));           // valid: all values >= 0
}

// ---------------------------------------------------------------------------
__global__ __launch_bounds__(256) void vq_gather(const float* __restrict__ latent,
                                                 const float* __restrict__ cbw,
                                                 float* __restrict__ out) {
  __shared__ float s[256];
  const int t = blockIdx.x * 256 + threadIdx.x;        // 0..VQ_ND-1
  const int n = t >> 7, d = t & 127;
  float q = cbw[g_idx[n] * VQ_D + d];                  // exact fp32 gather
  out[t] = q;                                          // quantized_st == quantized
  float diff = latent[t] - q;
  s[threadIdx.x] = diff * diff;
  __syncthreads();
  for (int off = 128; off; off >>= 1) { if (threadIdx.x < off) s[threadIdx.x] += s[threadIdx.x + off]; __syncthreads(); }
  if (threadIdx.x == 0) atomicAdd(&g_mse, s[0]);
}

__global__ __launch_bounds__(256) void vq_final(float* __restrict__ outs) {
  __shared__ float s[256];
  float e = 0.f;
  for (int k = threadIdx.x; k < VQ_K; k += 256) {
    float p = (float)g_counts[k] * (1.0f / (float)VQ_N);
    e -= p * logf(p + 1e-10f);
  }
  s[threadIdx.x] = e;
  __syncthreads();
  for (int off = 128; off; off >>= 1) { if (threadIdx.x < off) s[threadIdx.x] += s[threadIdx.x + off]; __syncthreads(); }
  if (threadIdx.x == 0) {
    float mse_mean = g_mse / (float)VQ_ND;
    outs[0] = 0.25f * mse_mean;                           // commitment_loss (BETA)
    outs[1] = mse_mean;                                   // codebook_loss
    outs[2] = expf(s[0]);                                 // perplexity
    outs[3] = g_selcos / (float)VQ_N;                     // sel_cos
    outs[4] = g_dsum / ((float)VQ_K * (float)(VQ_K - 1)); // avg_euclidean
    outs[5] = __uint_as_float(g_dmin);                    // min_euclidean
  }
}

// ---------------------------------------------------------------------------
extern "C" void kernel_launch(void* const* d_in, const int* in_sizes, int n_in,
                              void* d_out, int out_size, void* d_ws, size_t ws_size,
                              hipStream_t stream) {
  (void)in_sizes; (void)n_in; (void)out_size; (void)d_ws; (void)ws_size;
  const float* latent = (const float*)d_in[0];   // [16,1024,128] fp32
  const float* cbw    = (const float*)d_in[1];   // [4096,128]    fp32
  float* out = (float*)d_out;

  vq_init   <<<16,    256, 0, stream>>>();
  vq_prep_cb<<<VQ_K,  128, 0, stream>>>(cbw);
  vq_prep_x <<<VQ_N,  128, 0, stream>>>(latent);
  vq_scores <<<128,   128, 0, stream>>>(out + VQ_ND);          // idx region
  vq_gram   <<<256,   128, 0, stream>>>();
  vq_gather <<<VQ_ND / 256, 256, 0, stream>>>(latent, cbw, out);
  vq_final  <<<1,     256, 0, stream>>>(out + VQ_SCAL);
}
